// PlainAttnLayer_13348758356170
// MI455X (gfx1250) — matrix operand order repaired and checked by reading, not compiled, not run on hardware
//
#include <hip/hip_runtime.h>
#include <hip/hip_bf16.h>
#include <math.h>

// ---------------------------------------------------------------------------
// Types
// ---------------------------------------------------------------------------
typedef __bf16 bf16_t;
typedef __attribute__((ext_vector_type(16))) __bf16 v16bf;
typedef __attribute__((ext_vector_type(8)))  float  v8f;
typedef __attribute__((ext_vector_type(4)))  unsigned int u32x4;

union Frag16 {            // 16 bf16 = 32 bytes = two 16B chunks
    u32x4 u[2];
    v16bf v;
};

// Problem constants
constexpr int Bb    = 4;
constexpr int Ss    = 2048;
constexpr int Din   = 1024;
constexpr int Dhid  = 1024;
constexpr int Mrows = Bb * Ss;          // 8192

// GEMM tiling: 128x256 block tile, 8 wave32 in 2(M)x4(N), each wave 64x64
// (4x4 WMMA tiles -> 16 v_wmma per 32-wide K step vs 16 ds_load_b128)
constexpr int BM = 128;
constexpr int BN = 256;
constexpr int LDS_STRIDE = 40;          // bf16 elems; 80B row stride -> conflict-free frag reads

enum { EPI_BF16 = 0, EPI_GELU_BF16 = 1, EPI_MASK_F32 = 2, EPI_F32 = 3 };

// Async global->LDS DMA (CDNA5): per-lane 16B transfer, tracked by ASYNCcnt.
// INST_OFFSET is added to BOTH the LDS and the global address (ISA 08 §4.4).
__device__ __forceinline__ void async_b128(unsigned ldsOff, unsigned long long gaddr) {
    asm volatile("global_load_async_to_lds_b128 %0, %1, off"
                 :: "v"(ldsOff), "v"(gaddr) : "memory");
}
__device__ __forceinline__ void async_b128_o16(unsigned ldsOff, unsigned long long gaddr) {
    asm volatile("global_load_async_to_lds_b128 %0, %1, off offset:16"
                 :: "v"(ldsOff), "v"(gaddr) : "memory");
}
__device__ __forceinline__ void async_b128_o32(unsigned ldsOff, unsigned long long gaddr) {
    asm volatile("global_load_async_to_lds_b128 %0, %1, off offset:32"
                 :: "v"(ldsOff), "v"(gaddr) : "memory");
}
__device__ __forceinline__ void async_b128_o48(unsigned ldsOff, unsigned long long gaddr) {
    asm volatile("global_load_async_to_lds_b128 %0, %1, off offset:48"
                 :: "v"(ldsOff), "v"(gaddr) : "memory");
}
__device__ __forceinline__ void wait_asynccnt0() {
    asm volatile("s_wait_asynccnt 0x0" ::: "memory");
}

// ---------------------------------------------------------------------------
// fp32 -> bf16 elementwise convert
// ---------------------------------------------------------------------------
__global__ __launch_bounds__(256) void convert_f2b(const float* __restrict__ in,
                                                   bf16_t* __restrict__ out, size_t n) {
    size_t i = (size_t)blockIdx.x * blockDim.x + threadIdx.x;
    size_t stride = (size_t)gridDim.x * blockDim.x;
    for (; i < n; i += stride) out[i] = (bf16_t)in[i];
}

// ---------------------------------------------------------------------------
// Transpose (+convert to bf16).  in: [rows, cols] (T) -> out: [cols, rows] bf16
// ---------------------------------------------------------------------------
template <typename T>
__global__ __launch_bounds__(256) void transpose_to_bf16(const T* __restrict__ in,
                                                         bf16_t* __restrict__ out,
                                                         int rows, int cols,
                                                         size_t inBatchStride,
                                                         size_t outBatchStride) {
    __shared__ T tile[32][33];
    const size_t bo_in  = (size_t)blockIdx.z * inBatchStride;
    const size_t bo_out = (size_t)blockIdx.z * outBatchStride;
    const int c0 = blockIdx.x * 32;
    const int r0 = blockIdx.y * 32;
    const int tx = threadIdx.x;     // 0..31
    const int ty = threadIdx.y;     // 0..7
    for (int i = ty; i < 32; i += 8)
        tile[i][tx] = in[bo_in + (size_t)(r0 + i) * cols + (c0 + tx)];
    __syncthreads();
    for (int i = ty; i < 32; i += 8)
        out[bo_out + (size_t)(c0 + i) * rows + (r0 + tx)] = (bf16_t)(float)tile[tx][i];
}

// ---------------------------------------------------------------------------
// Batched NT GEMM:  C[M,N] = A[M,K] * B[N,K]^T   (bf16 in, fp32 accum via WMMA)
//   grid = (N/256, M/128, batch), block = 256 threads (8 wave32)
//   Double-buffered LDS filled by async global->LDS DMA; the DMA for tile k+1
//   runs concurrently with the 16 WMMAs of tile k.
// ---------------------------------------------------------------------------
template <int EPI>
__global__ __launch_bounds__(256) void gemm_bf16_wmma(
    const bf16_t* __restrict__ A, const bf16_t* __restrict__ Bm,
    int K, int lda, int ldb,
    size_t strideA, size_t strideB,
    float* __restrict__ Cf, bf16_t* __restrict__ Cb,
    int ldc, size_t strideC,
    const float* __restrict__ bias,
    const int* __restrict__ mask, size_t strideMask,
    float scale) {

    __shared__ alignas(16) bf16_t As[2][BM * LDS_STRIDE];   // 2 x 10 KB
    __shared__ alignas(16) bf16_t Bs[2][BN * LDS_STRIDE];   // 2 x 20 KB

    const int lane    = threadIdx.x & 31;
    const int wave    = threadIdx.x >> 5;
    const int waveM   = wave & 1;         // 0..1  -> 64 rows each
    const int waveN   = wave >> 1;        // 0..3  -> 64 cols each
    const int halfSel = lane >> 4;        // 0 | 1
    const int ln      = lane & 15;

    const int blockN = blockIdx.x * BN;
    const int blockM = blockIdx.y * BM;
    const size_t aOff = (size_t)blockIdx.z * strideA;
    const size_t bOff = (size_t)blockIdx.z * strideB;

    v8f acc[4][4] = {};

    // Staging assignments: A tile 128x32 (32B/thread), B tile 256x32 (64B/thread)
    const int aRow  = threadIdx.x >> 1;   // 0..127
    const int aHalf = threadIdx.x & 1;    // 16-elem K chunk
    const int bRow  = threadIdx.x;        // 0..255
    const bf16_t* gA = A + aOff + (size_t)(blockM + aRow) * lda + aHalf * 16;
    const bf16_t* gB = Bm + bOff + (size_t)(blockN + bRow) * ldb;

    // LDS byte offsets (flat shared pointer truncated to addr[31:0] = LDS offset)
    const unsigned ldsA0 = (unsigned)(size_t)&As[0][aRow * LDS_STRIDE + aHalf * 16];
    const unsigned ldsA1 = (unsigned)(size_t)&As[1][aRow * LDS_STRIDE + aHalf * 16];
    const unsigned ldsB0 = (unsigned)(size_t)&Bs[0][bRow * LDS_STRIDE];
    const unsigned ldsB1 = (unsigned)(size_t)&Bs[1][bRow * LDS_STRIDE];

    auto asyncStage = [&](int buf, int k0) {
        const unsigned la = buf ? ldsA1 : ldsA0;
        const unsigned lb = buf ? ldsB1 : ldsB0;
        const unsigned long long ga = (unsigned long long)(gA + k0);
        const unsigned long long gb = (unsigned long long)(gB + k0);
        async_b128(la, ga);
        async_b128_o16(la, ga);
        async_b128(lb, gb);
        async_b128_o16(lb, gb);
        async_b128_o32(lb, gb);
        async_b128_o48(lb, gb);
    };

    const int nk = K >> 5;                // K / 32
    asyncStage(0, 0);

    for (int kt = 0; kt < nk; ++kt) {
        const int cur = kt & 1;
        wait_asynccnt0();                 // our DMA writes to buf `cur` complete
        __syncthreads();                  // all waves' writes visible; prev reads of cur^1 done

        if (kt + 1 < nk) {
            asyncStage(cur ^ 1, (kt + 1) << 5);   // DMA next tile during the WMMAs
            if (kt + 2 < nk) {                    // pull tile kt+2 toward near cache
                __builtin_prefetch(gA + ((kt + 2) << 5), 0, 3);
                __builtin_prefetch(gB + ((kt + 2) << 5), 0, 3);
            }
        }

        // Fragments (ISA 16-bit A layout: lanes0-15 K=0-7/16-23, lanes16-31 K=8-15/24-31)
        Frag16 af[4], bfg[4];
#pragma unroll
        for (int ti = 0; ti < 4; ++ti) {
            const bf16_t* base = &As[cur][(waveM * 64 + ti * 16 + ln) * LDS_STRIDE];
            af[ti].u[0] = *(const u32x4*)(base + halfSel * 8);
            af[ti].u[1] = *(const u32x4*)(base + 16 + halfSel * 8);
        }
#pragma unroll
        for (int tj = 0; tj < 4; ++tj) {
            // B layout: lane = column, lanes0-15 hold K 0-7/16-23, lanes16-31 hold K 8-15/24-31
            const bf16_t* base = &Bs[cur][(waveN * 64 + tj * 16 + ln) * LDS_STRIDE];
            bfg[tj].u[0] = *(const u32x4*)(base + halfSel * 8);
            bfg[tj].u[1] = *(const u32x4*)(base + 16 + halfSel * 8);
        }
#pragma unroll
        for (int ti = 0; ti < 4; ++ti)
#pragma unroll
            for (int tj = 0; tj < 4; ++tj)
                acc[ti][tj] = __builtin_amdgcn_wmma_f32_16x16x32_bf16(
                    false, af[ti].v, false, bfg[tj].v,
                    (short)0, acc[ti][tj], false, false);
    }

    // Epilogue.  C f32 layout: VGPR v, lanes0-15 -> M=v, lanes16-31 -> M=v+8, N=lane%16
    const size_t cOff = (size_t)blockIdx.z * strideC;
    const size_t mOff = (size_t)blockIdx.z * strideMask;
#pragma unroll
    for (int ti = 0; ti < 4; ++ti) {
#pragma unroll
        for (int tj = 0; tj < 4; ++tj) {
            const int rowBase = blockM + waveM * 64 + ti * 16 + halfSel * 8;
            const int col     = blockN + waveN * 64 + tj * 16 + ln;
#pragma unroll
            for (int v = 0; v < 8; ++v) {
                const int row = rowBase + v;
                float val = acc[ti][tj][v];
                if constexpr (EPI == EPI_BF16 || EPI == EPI_GELU_BF16) {
                    if (bias) val += bias[col];
                    if constexpr (EPI == EPI_GELU_BF16)
                        val = 0.5f * val * (1.0f + erff(val * 0.70710678118654752f));
                    Cb[cOff + (size_t)row * ldc + col] = (bf16_t)val;
                } else if constexpr (EPI == EPI_MASK_F32) {
                    val = val * scale +
                          (float)mask[mOff + (size_t)row * ldc + col] * (-1.0e9f);
                    Cf[cOff + (size_t)row * ldc + col] = val;
                } else {
                    Cf[cOff + (size_t)row * ldc + col] = val;
                }
            }
        }
    }
}

// ---------------------------------------------------------------------------
// Row softmax over n=2048, in-place fp32 + bf16 copy.  One 256-thread block/row.
// ---------------------------------------------------------------------------
__global__ __launch_bounds__(256) void softmax_rows(float* __restrict__ scores,
                                                    bf16_t* __restrict__ probs,
                                                    int n) {
    const size_t row = blockIdx.x;
    float* p   = scores + row * (size_t)n;
    bf16_t* pb = probs + row * (size_t)n;
    const int lane = threadIdx.x & 31;
    const int wave = threadIdx.x >> 5;
    __shared__ float red[8];

    float vals[8];
    float m = -3.4e38f;
#pragma unroll
    for (int i = 0; i < 8; ++i) {
        vals[i] = p[threadIdx.x + i * 256];
        m = fmaxf(m, vals[i]);
    }
#pragma unroll
    for (int o = 16; o > 0; o >>= 1) m = fmaxf(m, __shfl_xor(m, o, 32));
    if (lane == 0) red[wave] = m;
    __syncthreads();
    m = red[0];
#pragma unroll
    for (int i = 1; i < 8; ++i) m = fmaxf(m, red[i]);
    __syncthreads();

    float s = 0.0f;
#pragma unroll
    for (int i = 0; i < 8; ++i) {
        vals[i] = expf(vals[i] - m);
        s += vals[i];
    }
#pragma unroll
    for (int o = 16; o > 0; o >>= 1) s += __shfl_xor(s, o, 32);
    if (lane == 0) red[wave] = s;
    __syncthreads();
    s = red[0];
#pragma unroll
    for (int i = 1; i < 8; ++i) s += red[i];
    const float inv = 1.0f / s;
#pragma unroll
    for (int i = 0; i < 8; ++i) {
        const float o = vals[i] * inv;
        p[threadIdx.x + i * 256]  = o;
        pb[threadIdx.x + i * 256] = (bf16_t)o;
    }
}

// ---------------------------------------------------------------------------
// Orchestration
// ---------------------------------------------------------------------------
extern "C" void kernel_launch(void* const* d_in, const int* in_sizes, int n_in,
                              void* d_out, int out_size, void* d_ws, size_t ws_size,
                              hipStream_t stream) {
    (void)in_sizes; (void)n_in; (void)out_size; (void)ws_size;

    const float* inputs = (const float*)d_in[0];
    const int*   mask   = (const int*)d_in[1];
    const float* Wq     = (const float*)d_in[2];
    const float* bq     = (const float*)d_in[3];
    const float* Wk     = (const float*)d_in[4];
    const float* bk     = (const float*)d_in[5];
    const float* Wv     = (const float*)d_in[6];
    const float* bv     = (const float*)d_in[7];

    float* out_ctx  = (float*)d_out;                         // [4,2048,1024]
    float* out_attn = out_ctx + (size_t)Bb * Ss * Dhid;      // [4,2048,2048]

    // Workspace layout (bf16, ~118 MB)
    bf16_t* Xbf = (bf16_t*)d_ws;                             // [8192,1024]
    bf16_t* Wqt = Xbf + (size_t)Mrows * Din;                 // [1024,1024] (N,K)
    bf16_t* Wkt = Wqt + (size_t)Din * Din;
    bf16_t* Wvt = Wkt + (size_t)Din * Din;
    bf16_t* Qbf = Wvt + (size_t)Din * Din;                   // [8192,1024]
    bf16_t* Kbf = Qbf + (size_t)Mrows * Din;                 // [8192,1024]
    bf16_t* Vbf = Kbf + (size_t)Mrows * Din;                 // gelu(XWv) [8192,1024]
    bf16_t* Vt  = Vbf + (size_t)Mrows * Dhid;                // per-batch [1024,2048]
    bf16_t* Pbf = Vt  + (size_t)Mrows * Dhid;                // attn bf16 [4,2048,2048]

    const dim3 blk256(256), blkT(32, 8);

    // 1) Precision staging
    convert_f2b<<<dim3(4096), blk256, 0, stream>>>(inputs, Xbf, (size_t)Mrows * Din);
    transpose_to_bf16<float><<<dim3(32, 32, 1), blkT, 0, stream>>>(Wq, Wqt, Din, Din, 0, 0);
    transpose_to_bf16<float><<<dim3(32, 32, 1), blkT, 0, stream>>>(Wk, Wkt, Din, Din, 0, 0);
    transpose_to_bf16<float><<<dim3(32, 32, 1), blkT, 0, stream>>>(Wv, Wvt, Din, Din, 0, 0);

    // 2) Projections: Q, K (bias), WV (bias + erf-GELU), all 8192x1024x1024 NT
    gemm_bf16_wmma<EPI_BF16><<<dim3(4, 64, 1), blk256, 0, stream>>>(
        Xbf, Wqt, Din, Din, Din, 0, 0, nullptr, Qbf, Din, 0, bq, nullptr, 0, 1.0f);
    gemm_bf16_wmma<EPI_BF16><<<dim3(4, 64, 1), blk256, 0, stream>>>(
        Xbf, Wkt, Din, Din, Din, 0, 0, nullptr, Kbf, Din, 0, bk, nullptr, 0, 1.0f);
    gemm_bf16_wmma<EPI_GELU_BF16><<<dim3(4, 64, 1), blk256, 0, stream>>>(
        Xbf, Wvt, Din, Din, Din, 0, 0, nullptr, Vbf, Din, 0, bv, nullptr, 0, 1.0f);

    // 3) Transpose WV per batch for NT context GEMM: [2048,1024] -> [1024,2048]
    transpose_to_bf16<bf16_t><<<dim3(32, 64, Bb), blkT, 0, stream>>>(
        Vbf, Vt, Ss, Dhid, (size_t)Ss * Dhid, (size_t)Dhid * Ss);

    // 4) Scores: Q @ K^T * (1/32) + mask*(-1e9) -> fp32 into attn output region
    gemm_bf16_wmma<EPI_MASK_F32><<<dim3(8, 16, Bb), blk256, 0, stream>>>(
        Qbf, Kbf, Din, Din, Din, (size_t)Ss * Din, (size_t)Ss * Din,
        out_attn, nullptr, Ss, (size_t)Ss * Ss, nullptr, mask, (size_t)Ss * Ss,
        0.03125f);

    // 5) Softmax in place (fp32 out) + bf16 probs for context GEMM
    softmax_rows<<<dim3(Bb * Ss), blk256, 0, stream>>>(out_attn, Pbf, Ss);

    // 6) Context: attn[2048,2048] @ WV[2048,1024] (NT with Vt), fp32 out
    gemm_bf16_wmma<EPI_F32><<<dim3(4, 16, Bb), blk256, 0, stream>>>(
        Pbf, Vt, Ss, Ss, Ss, (size_t)Ss * Ss, (size_t)Dhid * Ss,
        out_ctx, nullptr, Dhid, (size_t)Ss * Dhid, nullptr, nullptr, 0, 1.0f);
}
